// PatchInpainting_40810779247137
// MI455X (gfx1250) — matrix operand order repaired and checked by reading, not compiled, not run on hardware
//
#include <hip/hip_runtime.h>
#include <math.h>

// ---------------- constants (match reference) ----------------
#define BB    2
#define CC    3
#define HH    512
#define WW    512
#define KK    8
#define GG    64
#define LL    4096          // G*G
#define FD    128
#define PD    192           // C*K*K
#define MD    320           // PD+FD
#define DDIM  256
#define TOPK  16
#define RH    128
#define K1P   288           // W1 K padded to multiple of 32 (from 259)
#define NEGV  (-1.0e9f)

typedef __attribute__((ext_vector_type(16))) __bf16 v16bf;
typedef __attribute__((ext_vector_type(8)))  float  v8f;

// ---------------- WMMA helpers (CDNA5 bf16 16x16x32) ----------------
__device__ __forceinline__ v8f wmma_bf16(v16bf a, v16bf b, v8f c) {
  // 8 args: (neg_a, A, neg_b, B, c_mod, C, reuse_a, reuse_b)
  return __builtin_amdgcn_wmma_f32_16x16x32_bf16(false, a, false, b, (short)0, c,
                                                 false, false);
}

// A-matrix (16x32, row-major source, lda elements): ISA 16-bit A layout.
// Per lane: two contiguous 16-byte chunks -> 2x global_load_b128.
__device__ __forceinline__ v16bf load_a_frag(const __bf16* A, int lda,
                                             int row_base, int k0, int lane) {
  int m  = row_base + (lane & 15);
  int hf = (lane >> 4) & 1;
  const __bf16* rp = A + (size_t)m * lda + k0;
  v16bf a;
#pragma unroll
  for (int j = 0; j < 16; ++j) {
    int r = j >> 1;
    int k = ((r < 4) ? 0 : 16) + hf * 8 + (r & 3) * 2 + (j & 1);
    a[j] = rp[k];
  }
  return a;
}

// B-matrix (32x16) where columns are stored as contiguous rows of length ld
// (i.e. source is N x K row-major, or keysᵀ): two 8-bf16 chunks per lane (k 8h.., 16+8h..).
__device__ __forceinline__ v16bf load_b_frag_rows(const __bf16* Rows, int ld,
                                                  int row_base, int k0, int lane) {
  int n  = (lane & 15);
  int hf = (lane >> 4) & 1;
  const __bf16* rp = Rows + (size_t)(row_base + n) * ld + k0 + hf * 8;
  v16bf b;
#pragma unroll
  for (int j = 0; j < 16; ++j) b[j] = rp[(j < 8) ? j : j + 8];
  return b;
}

// ---------------- K0a: tokenize image + features -> match (f32 + bf16) ------
__global__ void __launch_bounds__(256)
build_match(const float* __restrict__ image, const float* __restrict__ features,
            float* __restrict__ mf, __bf16* __restrict__ mb) {
  size_t i = (size_t)blockIdx.x * blockDim.x + threadIdx.x;
  if (i >= (size_t)BB * LL * MD) return;
  int p = (int)(i % MD);
  size_t bl = i / MD;
  int l = (int)(bl % LL);
  int b = (int)(bl / LL);
  int gy = l >> 6, gx = l & 63;
  float v;
  if (p < PD) {
    int c = p >> 6, rem = p & 63, ky = rem >> 3, kx = rem & 7;
    v = image[(((size_t)(b * CC + c) * HH + gy * KK + ky) * WW + gx * KK + kx)];
  } else {
    int f = p - PD;
    v = features[(((size_t)(b * FD + f) * GG + gy) * GG + gx)];
  }
  mf[i] = v;
  mb[i] = (__bf16)v;
}

// ---------------- K0b: weights -> bf16, TRANSPOSED to N x K ----------------
// Transposed layout makes every WMMA B-fragment two contiguous b128 loads.
__global__ void __launch_bounds__(256)
conv_weights(const float* __restrict__ Wd, const float* __restrict__ Wq,
             const float* __restrict__ Wk, const float* __restrict__ W1,
             __bf16* __restrict__ WdT, __bf16* __restrict__ WqT,
             __bf16* __restrict__ WkT, __bf16* __restrict__ W1T) {
  int i = blockIdx.x * blockDim.x + threadIdx.x;
  if (i < MD * DDIM) {                 // Wd: (MD x DDIM) -> WdT: (DDIM x MD)
    int k = i / DDIM, n = i % DDIM;
    WdT[(size_t)n * MD + k] = (__bf16)Wd[i];
  }
  if (i < MD * RH) {                   // Wq/Wk: (MD x RH) -> (RH x MD)
    int k = i / RH, n = i % RH;
    WqT[(size_t)n * MD + k] = (__bf16)Wq[i];
    WkT[(size_t)n * MD + k] = (__bf16)Wk[i];
  }
  if (i < RH * K1P) {                  // W1: (259 x RH) -> padded (RH x 288)
    int n = i / K1P, k = i % K1P;
    W1T[i] = (k < 2 * RH + 3) ? (__bf16)W1[(size_t)k * RH + n] : (__bf16)0.0f;
  }
}

// ---------------- K1: bf16 WMMA GEMM, B pre-transposed (N x K) --------------
// One wave per 16x16 tile; compile-time K so the 10-step k-loop fully unrolls.
template <int KD, bool WF, bool WB>
__global__ void __launch_bounds__(256)
gemm_bf16(const __bf16* __restrict__ A, const __bf16* __restrict__ Bt,
          float* __restrict__ Cf, __bf16* __restrict__ Cb,
          int M, int N, float scale) {
  int wid  = (blockIdx.x * blockDim.x + threadIdx.x) >> 5;
  int lane = threadIdx.x & 31;
  int ntn  = N >> 4;
  int total = (M >> 4) * ntn;
  if (wid >= total) return;
  int tm = wid / ntn, tn = wid % ntn;
  v8f c = {};
#pragma unroll
  for (int k0 = 0; k0 < KD; k0 += 32) {
    v16bf a = load_a_frag(A, KD, tm * 16, k0, lane);
    v16bf b = load_b_frag_rows(Bt, KD, tn * 16, k0, lane);
    c = wmma_bf16(a, b, c);
  }
  int hf = (lane >> 4) & 1;
  size_t base = (size_t)(tm * 16 + hf * 8) * N + tn * 16 + (lane & 15);
#pragma unroll
  for (int r = 0; r < 8; ++r) {
    float v = c[r] * scale;
    if (WF) Cf[base + (size_t)r * N] = v;
    if (WB) Cb[base + (size_t)r * N] = (__bf16)v;
  }
}

// ---------------- K2: logits (desc @ descT) via WMMA + masked top-16 --------
// One block per 16 queries; 8 waves; 16x4096 f32 logit strip in 256 KB LDS.
__global__ void __launch_bounds__(256)
logits_topk(const __bf16* __restrict__ desc, const int* __restrict__ mask,
            float* __restrict__ topv, int* __restrict__ topi) {
  extern __shared__ float lds[];                  // 16 * 4096 floats
  int b    = blockIdx.x >> 8;                     // 256 query tiles per batch
  int qt   = blockIdx.x & 255;
  int tid  = threadIdx.x;
  int wave = tid >> 5, lane = tid & 31;
  int hf   = (lane >> 4) & 1;
  int qrow = b * LL + qt * 16;

  v16bf aF[8];
#pragma unroll
  for (int ks = 0; ks < 8; ++ks)
    aF[ks] = load_a_frag(desc, DDIM, qrow, ks * 32, lane);

  for (int kt = wave; kt < 256; kt += 8) {
    v8f c = {};
    int krow = b * LL + kt * 16;
#pragma unroll
    for (int ks = 0; ks < 8; ++ks) {
      v16bf bfrag = load_b_frag_rows(desc, DDIM, krow, ks * 32, lane);
      c = wmma_bf16(aF[ks], bfrag, c);
    }
    int n   = lane & 15;
    int key = kt * 16 + n;
    float pen = mask[b * LL + key] ? NEGV : 0.0f;
#pragma unroll
    for (int r = 0; r < 8; ++r)
      lds[(r + hf * 8) * LL + key] = c[r] + pen;
  }
  __syncthreads();

  // top-16 per query row: one wave per row, 2 rows per wave; iterative argmax.
  for (int mi = 0; mi < 2; ++mi) {
    int m = wave + mi * 8;
    float* rowp = lds + (size_t)m * LL;
    int gq = b * LL + qt * 16 + m;
    for (int round = 0; round < TOPK; ++round) {
      float v = -INFINITY; int bi = 0x7fffffff;
      for (int i = lane; i < LL; i += 32) {
        float f = rowp[i];
        if (f > v || (f == v && i < bi)) { v = f; bi = i; }
      }
#pragma unroll
      for (int off = 16; off >= 1; off >>= 1) {
        float ov = __shfl_xor(v, off, 32);
        int   oi = __shfl_xor(bi, off, 32);
        if (ov > v || (ov == v && oi < bi)) { v = ov; bi = oi; }
      }
      if (lane == 0) {
        topv[(size_t)gq * TOPK + round] = v;
        topi[(size_t)gq * TOPK + round] = bi;
        rowp[bi] = -INFINITY;
      }
      __syncthreads();
    }
  }
}

// ---------------- K3: refinement MLP (WMMA) + softmax + output --------------
// One wave per token. Masked token: full pipeline; else copy input pixels.
__global__ void __launch_bounds__(32)
refine_out(const float* __restrict__ matchf, const float* __restrict__ qf,
           const float* __restrict__ kf, const float* __restrict__ topv,
           const int* __restrict__ topi, const int* __restrict__ mask,
           const __bf16* __restrict__ W1T, const float* __restrict__ b1,
           const float* __restrict__ W2, const float* __restrict__ b2,
           const float* __restrict__ image, float* __restrict__ out) {
  int b = blockIdx.x >> 12;          // L = 4096
  int l = blockIdx.x & (LL - 1);
  int lane = threadIdx.x;
  int gy = l >> 6, gx = l & 63;

  if (!mask[b * LL + l]) {
    for (int p = lane; p < PD; p += 32) {
      int c = p >> 6, rem = p & 63, ky = rem >> 3, kx = rem & 7;
      size_t off = (((size_t)(b * CC + c) * HH + gy * KK + ky) * WW + gx * KK + kx);
      out[off] = image[off];
    }
    return;
  }

  __shared__ __bf16 feats[16 * K1P];
  __shared__ float  sref[16];
  __shared__ int    sidx[16];
  __shared__ float  stl[16];

  if (lane < TOPK) {
    sidx[lane] = topi[(size_t)(b * LL + l) * TOPK + lane];
    stl[lane]  = topv[(size_t)(b * LL + l) * TOPK + lane];
  }
  __syncthreads();

  const float* qrow = qf + (size_t)(b * LL + l) * RH;
  float pqy = (float)gy / (float)GG, pqx = (float)gx / (float)GG;
  for (int m = 0; m < TOPK; ++m) {
    int idx = sidx[m];
    const float* krow = kf + (size_t)(b * LL + idx) * RH;
    for (int col = lane; col < K1P; col += 32) {
      float v;
      if      (col < RH)       v = qrow[col] * krow[col];
      else if (col < 2 * RH)   v = fabsf(qrow[col - RH] - krow[col - RH]);
      else if (col == 2 * RH)  v = stl[m];
      else if (col == 2*RH+1)  v = (float)(idx >> 6) / (float)GG - pqy;
      else if (col == 2*RH+2)  v = (float)(idx & 63) / (float)GG - pqx;
      else                     v = 0.0f;
      feats[m * K1P + col] = (__bf16)v;
    }
  }
  __syncthreads();

  int hf = (lane >> 4) & 1;
  int n16 = lane & 15;
  v16bf aF[9];
#pragma unroll
  for (int ks = 0; ks < 9; ++ks)
    aF[ks] = load_a_frag(feats, K1P, 0, ks * 32, lane);

  float partial[8];
#pragma unroll
  for (int r = 0; r < 8; ++r) partial[r] = 0.0f;

  for (int t = 0; t < 8; ++t) {                  // 8 n-tiles of 16 -> RH=128
    v8f acc = {};
#pragma unroll
    for (int ks = 0; ks < 9; ++ks) {
      v16bf bfrag = load_b_frag_rows(W1T, K1P, t * 16, ks * 32, lane);
      acc = wmma_bf16(aF[ks], bfrag, acc);
    }
    int col = t * 16 + n16;
    float bb = b1[col], w2v = W2[col];
#pragma unroll
    for (int r = 0; r < 8; ++r) {
      float h = acc[r] + bb;
      float g = 0.5f * h * (1.0f + erff(h * 0.70710678118654752f)); // exact GELU
      partial[r] += g * w2v;
    }
  }
  // sum over 16 columns held within each lane-half
#pragma unroll
  for (int r = 0; r < 8; ++r) {
#pragma unroll
    for (int off = 8; off >= 1; off >>= 1)
      partial[r] += __shfl_xor(partial[r], off, 32);
  }
  if (n16 == 0) {
#pragma unroll
    for (int r = 0; r < 8; ++r) {
      int m = r + hf * 8;
      sref[m] = stl[m] + partial[r] + b2[0];
    }
  }
  __syncthreads();

  // softmax over 16 refined scores (all lanes redundantly)
  float wts[16];
  float mx = -INFINITY;
#pragma unroll
  for (int m = 0; m < TOPK; ++m) mx = fmaxf(mx, sref[m]);
  float s = 0.0f;
#pragma unroll
  for (int m = 0; m < TOPK; ++m) { wts[m] = __expf(sref[m] - mx); s += wts[m]; }
  float inv = 1.0f / s;

  for (int p = lane; p < PD; p += 32) {
    float acc = 0.0f;
#pragma unroll
    for (int m = 0; m < TOPK; ++m)
      acc += wts[m] * matchf[(size_t)(b * LL + sidx[m]) * MD + p];
    acc *= inv;
    int c = p >> 6, rem = p & 63, ky = rem >> 3, kx = rem & 7;
    size_t off = (((size_t)(b * CC + c) * HH + gy * KK + ky) * WW + gx * KK + kx);
    out[off] = acc;
  }
}

// ---------------- host launcher ----------------
extern "C" void kernel_launch(void* const* d_in, const int* in_sizes, int n_in,
                              void* d_out, int out_size, void* d_ws, size_t ws_size,
                              hipStream_t stream) {
  const float* image    = (const float*)d_in[0];
  const float* features = (const float*)d_in[1];
  const int*   tmask    = (const int*)  d_in[2];
  const float* W_desc   = (const float*)d_in[3];
  const float* Wq_rr    = (const float*)d_in[4];
  const float* Wk_rr    = (const float*)d_in[5];
  const float* W1       = (const float*)d_in[6];
  const float* b1       = (const float*)d_in[7];
  const float* W2       = (const float*)d_in[8];
  const float* b2       = (const float*)d_in[9];
  float* out = (float*)d_out;

  char* ws = (char*)d_ws;
  size_t off = 0;
  auto take = [&](size_t bytes) -> char* {
    char* p = ws + off;
    off = (off + bytes + 255) & ~(size_t)255;
    return p;
  };
  const size_t ROWS = (size_t)BB * LL;           // 8192
  float*  match_f  = (float*) take(ROWS * MD * 4);
  __bf16* match_b  = (__bf16*)take(ROWS * MD * 2);
  __bf16* desc_b   = (__bf16*)take(ROWS * DDIM * 2);
  float*  qf       = (float*) take(ROWS * RH * 4);
  float*  kf       = (float*) take(ROWS * RH * 4);
  float*  topv     = (float*) take(ROWS * TOPK * 4);
  int*    topi     = (int*)   take(ROWS * TOPK * 4);
  __bf16* WdT      = (__bf16*)take((size_t)DDIM * MD * 2);
  __bf16* WqT      = (__bf16*)take((size_t)RH * MD * 2);
  __bf16* WkT      = (__bf16*)take((size_t)RH * MD * 2);
  __bf16* W1T      = (__bf16*)take((size_t)RH * K1P * 2);

  // K0a: tokenize
  {
    size_t total = ROWS * MD;
    build_match<<<(unsigned)((total + 255) / 256), 256, 0, stream>>>(
        image, features, match_f, match_b);
  }
  // K0b: weight conversion to transposed bf16 (+ W1 zero-pad to K=288)
  conv_weights<<<(MD * DDIM + 255) / 256, 256, 0, stream>>>(
      W_desc, Wq_rr, Wk_rr, W1, WdT, WqT, WkT, W1T);

  // K1: desc = (match @ W_desc) * 0.25  -> bf16  (scale folds 1/sqrt(DD))
  {
    int tiles = (int)(ROWS / 16) * (DDIM / 16);       // 8192 waves
    gemm_bf16<MD, false, true><<<(tiles * 32 + 255) / 256, 256, 0, stream>>>(
        match_b, WdT, (float*)nullptr, desc_b, (int)ROWS, DDIM, 0.25f);
  }
  // K1: qf = match @ Wq_rr ; kf = match @ Wk_rr  (f32)
  {
    int tiles = (int)(ROWS / 16) * (RH / 16);         // 4096 waves
    gemm_bf16<MD, true, false><<<(tiles * 32 + 255) / 256, 256, 0, stream>>>(
        match_b, WqT, qf, (__bf16*)nullptr, (int)ROWS, RH, 1.0f);
    gemm_bf16<MD, true, false><<<(tiles * 32 + 255) / 256, 256, 0, stream>>>(
        match_b, WkT, kf, (__bf16*)nullptr, (int)ROWS, RH, 1.0f);
  }
  // K2: logits + top-16 (256 KB dynamic LDS per workgroup; 320 KB/WGP)
  {
    const size_t ldsBytes = (size_t)16 * LL * sizeof(float);  // 262144
    hipFuncSetAttribute((const void*)logits_topk,
                        hipFuncAttributeMaxDynamicSharedMemorySize,
                        (int)ldsBytes);
    logits_topk<<<BB * (LL / 16), 256, ldsBytes, stream>>>(
        desc_b, tmask, topv, topi);
  }
  // K3: refinement MLP + softmax + output scatter
  refine_out<<<BB * LL, 32, 0, stream>>>(
      match_f, qf, kf, topv, topi, tmask, W1T, b1, W2, b2, image, out);
}
